// Gemma4DecoderWrapper_62921270886443
// MI455X (gfx1250) — compile-verified, matrix operand order checked by reading, not run on hardware
//
#include <hip/hip_runtime.h>
#include <hip/hip_bf16.h>
#include <math.h>

// ---------------------------------------------------------------------------
// Gemma-style 4-layer decode step for MI455X (gfx1250, wave32).
// Bandwidth-bound (~1.3 GB weight traffic, ~0.7 GFLOP): GEMVs are K-split
// streaming kernels with deterministic fixed-order reduction; attention uses
// v_wmma_f32_16x16x32_f16 (q heads padded to the 16-row WMMA tile) with
// vectorized b128 fragment loads and branch-free zero padding.
// ---------------------------------------------------------------------------

typedef __attribute__((ext_vector_type(16))) _Float16 v16h;
typedef __attribute__((ext_vector_type(8)))  _Float16 v8h;
typedef __attribute__((ext_vector_type(8)))  float    v8f;
typedef __attribute__((ext_vector_type(4)))  float    v4f;

#define LAYERS  4
#define HDIM    2048
#define NH      8
#define NKV     2
#define NREP    (NH / NKV)       // 4
#define HD      256
#define FF      8192
#define VOCAB   32000
#define CTX     4096
#define PLD     256
#define SPLITK  8
#define JSPLIT  8
#define BLK     256
#define SOFTCAPF 30.0f

__device__ __forceinline__ float gelu_tanh(float x) {
    const float c = 0.7978845608028654f; // sqrt(2/pi)
    float t = c * (x + 0.044715f * x * x * x);
    return 0.5f * x * (1.0f + tanhf(t));
}

// ---------------------------------------------------------------- rmsnorm ---
__global__ void rmsnorm_kernel(const float* __restrict__ x,
                               const float* __restrict__ w,
                               float* __restrict__ y, int n) {
    __shared__ float red[BLK];
    float s = 0.f;
    for (int i = threadIdx.x; i < n; i += BLK) { float v = x[i]; s += v * v; }
    red[threadIdx.x] = s; __syncthreads();
    for (int o = BLK / 2; o > 0; o >>= 1) {
        if (threadIdx.x < o) red[threadIdx.x] += red[threadIdx.x + o];
        __syncthreads();
    }
    float r = rsqrtf(red[0] / (float)n + 1e-6f);
    for (int i = threadIdx.x; i < n; i += BLK) y[i] = x[i] * r * w[i];
}

// h_out = (resid + rms(o)*w) * scale   (resid may alias h_out)
__global__ void rms_add_kernel(const float* __restrict__ o,
                               const float* __restrict__ w,
                               const float* __restrict__ resid,
                               float* __restrict__ hout, int n,
                               const float* __restrict__ ls, int lidx) {
    __shared__ float red[BLK];
    float s = 0.f;
    for (int i = threadIdx.x; i < n; i += BLK) { float v = o[i]; s += v * v; }
    red[threadIdx.x] = s; __syncthreads();
    for (int oo = BLK / 2; oo > 0; oo >>= 1) {
        if (threadIdx.x < oo) red[threadIdx.x] += red[threadIdx.x + oo];
        __syncthreads();
    }
    float r = rsqrtf(red[0] / (float)n + 1e-6f);
    float sc = (ls != nullptr) ? ls[lidx] : 1.0f;
    for (int i = threadIdx.x; i < n; i += BLK)
        hout[i] = (resid[i] + o[i] * r * w[i]) * sc;
}

// ------------------------------------------------------------------- gemv ---
// y[n] = sum_k x[k] * W[k*N + n], split over K into SPLITK deterministic
// partials (grid.y), x chunk staged in LDS, W streamed coalesced once.
__global__ void gemv_part_kernel(const float* __restrict__ x,
                                 const float* __restrict__ W,
                                 float* __restrict__ part, int K, int N) {
    __shared__ float xs[FF / SPLITK]; // 1024 floats max
    const int seg = blockIdx.y;
    const int klen = K / SPLITK;
    const int k0 = seg * klen;
    for (int i = threadIdx.x; i < klen; i += BLK) xs[i] = x[k0 + i];
    __syncthreads();
    const int n = blockIdx.x * BLK + threadIdx.x;
    if (n >= N) return;
    const float* Wp = W + (size_t)k0 * N + n;
    float acc = 0.f;
#pragma unroll 4
    for (int k = 0; k < klen; ++k) acc += xs[k] * Wp[(size_t)k * N];
    part[(size_t)seg * N + n] = acc;
}

// y[n] = sum over NSEG fixed-order partial segments (also reused to reduce
// the J-split attention partials).
__global__ void gemv_reduce_kernel(const float* __restrict__ part,
                                   float* __restrict__ y, int N) {
    const int n = blockIdx.x * BLK + threadIdx.x;
    if (n >= N) return;
    float a = 0.f;
#pragma unroll
    for (int s = 0; s < SPLITK; ++s) a += part[(size_t)s * N + n];
    y[n] = a;
}

// --------------------------------------------- q/k/v per-head rms + rope ---
// blocks: 0..7 q heads, 8..9 k heads, 10..11 v heads; 256 threads = HD dims.
__global__ void qkv_post_kernel(const float* __restrict__ qbuf,
                                const float* __restrict__ kbuf,
                                const float* __restrict__ vbuf,
                                const float* __restrict__ wq_norm,
                                const float* __restrict__ wk_norm,
                                const float* __restrict__ cosT,
                                const float* __restrict__ sinT,
                                const int* __restrict__ posp,
                                _Float16* __restrict__ qh,
                                float* __restrict__ k_new,
                                float* __restrict__ v_new) {
    const int b = blockIdx.x;
    const int d = threadIdx.x;   // HD == BLK
    __shared__ float xs[HD];
    __shared__ float red[HD];
    const float* src; const float* w = nullptr; int mode;
    if (b < NH)            { src = qbuf + b * HD;              w = wq_norm; mode = 0; }
    else if (b < NH + NKV) { src = kbuf + (b - NH) * HD;       w = wk_norm; mode = 1; }
    else                   { src = vbuf + (b - NH - NKV) * HD;              mode = 2; }
    float xv = src[d];
    red[d] = xv * xv; __syncthreads();
    for (int o = HD / 2; o > 0; o >>= 1) {
        if (d < o) red[d] += red[d + o];
        __syncthreads();
    }
    float r = rsqrtf(red[0] / (float)HD + 1e-6f);
    float y = xv * r * (mode == 2 ? 1.0f : w[d]);
    if (mode == 2) { v_new[(b - NH - NKV) * HD + d] = y; return; }
    xs[d] = y; __syncthreads();
    const int pos = posp[0];
    float rot = (d < HD / 2) ? -xs[d + HD / 2] : xs[d - HD / 2];
    float c = cosT[(size_t)pos * HD + d];
    float s = sinT[(size_t)pos * HD + d];
    float out = y * c + rot * s;
    if (mode == 0) qh[b * HD + d] = (_Float16)out;
    else           k_new[(b - NH) * HD + d] = out;
}

// ------------------------------------------------- attention scores (WMMA) --
// grid (CTX/16, NKV), 32 threads (one wave). A = padded q heads (16x32 f16,
// rows 4..15 zeroed via mask-multiply), B = K-cache tile (32x16, b128 loads +
// cvt_pk_f16), accumulate f32 over HD, 8 wmma per tile.
__global__ void attn_scores_kernel(const _Float16* __restrict__ qh,
                                   const float* __restrict__ Kcache, // [NKV][CTX][HD]
                                   const float* __restrict__ k_new,  // [NKV*HD]
                                   const float* __restrict__ mask,   // [CTX]
                                   const int* __restrict__ posp,
                                   float* __restrict__ scores) {     // [NH*CTX]
    const int lane = threadIdx.x;
    const int g    = blockIdx.y;
    const int j0   = blockIdx.x * 16;
    const int hi   = lane >> 4;
    const int mrow = lane & 15;
    const int j    = j0 + mrow;            // this lane's B column / C column
    const int pos  = posp[0];
    const float* Kp = (j == pos) ? (k_new + g * HD)
                                 : (Kcache + ((size_t)g * CTX + j) * HD);
    // A row: wrap to a valid row, zero via mask (branch-free padding).
    const _Float16* qrow = qh + (size_t)(g * NREP + (mrow & (NREP - 1))) * HD;
    const _Float16 rm = (mrow < NREP) ? (_Float16)1.0f : (_Float16)0.0f;
    v8f c = {};
    for (int k0 = 0; k0 < HD; k0 += 32) {
        const int base = k0 + hi * 8;
        v8h alo = *(const v8h*)(qrow + base) * rm;
        v8h ahi = *(const v8h*)(qrow + base + 16) * rm;
        v4f f0 = *(const v4f*)(Kp + base);
        v4f f1 = *(const v4f*)(Kp + base + 4);
        v4f f2 = *(const v4f*)(Kp + base + 16);
        v4f f3 = *(const v4f*)(Kp + base + 20);
        v16h a, b;
#pragma unroll
        for (int e = 0; e < 8; ++e) { a[e] = alo[e]; a[8 + e] = ahi[e]; }
#pragma unroll
        for (int e = 0; e < 4; ++e) {
            b[e]      = (_Float16)f0[e];
            b[4 + e]  = (_Float16)f1[e];
            b[8 + e]  = (_Float16)f2[e];
            b[12 + e] = (_Float16)f3[e];
        }
        c = __builtin_amdgcn_wmma_f32_16x16x32_f16(false, a, false, b,
                                                   (short)0, c, false, false);
    }
    const float scale = 0.0625f; // 1/sqrt(256)
    const float m = mask[j];
#pragma unroll
    for (int r = 0; r < 8; ++r) {
        const int M = r + 8 * hi;
        if (M < NREP) scores[(size_t)(g * NREP + M) * CTX + j] = c[r] * scale + m;
    }
}

// ----------------------------------------------------------------- softmax --
__global__ void softmax_kernel(const float* __restrict__ scores,
                               _Float16* __restrict__ probs) {
    const int h = blockIdx.x;
    const float* s = scores + (size_t)h * CTX;
    __shared__ float red[BLK];
    float m = -1e30f;
    for (int i = threadIdx.x; i < CTX; i += BLK) m = fmaxf(m, s[i]);
    red[threadIdx.x] = m; __syncthreads();
    for (int o = BLK / 2; o > 0; o >>= 1) {
        if (threadIdx.x < o) red[threadIdx.x] = fmaxf(red[threadIdx.x], red[threadIdx.x + o]);
        __syncthreads();
    }
    m = red[0]; __syncthreads();
    float sum = 0.f;
    for (int i = threadIdx.x; i < CTX; i += BLK) sum += expf(s[i] - m);
    red[threadIdx.x] = sum; __syncthreads();
    for (int o = BLK / 2; o > 0; o >>= 1) {
        if (threadIdx.x < o) red[threadIdx.x] += red[threadIdx.x + o];
        __syncthreads();
    }
    const float inv = 1.0f / red[0];
    for (int i = threadIdx.x; i < CTX; i += BLK)
        probs[(size_t)h * CTX + i] = (_Float16)(expf(s[i] - m) * inv);
}

// -------------------------------------------------- attention P·V (WMMA) ---
// grid (HD/16, NKV, JSPLIT), 32 threads. A = padded probs (16x32 f16, b128
// loads + mask-multiply), B = V tile (32 positions x 16 dims; each element's
// load is wave-coalesced across the 16 consecutive d lanes). CTX is split
// 8-ways into deterministic partial tiles reduced afterwards.
__global__ void attn_av_kernel(const _Float16* __restrict__ probs,
                               const float* __restrict__ Vcache, // [NKV][CTX][HD]
                               const float* __restrict__ v_new,  // [NKV*HD]
                               const int* __restrict__ posp,
                               float* __restrict__ attn_part) {  // [JSPLIT][NH*HD]
    const int lane = threadIdx.x;
    const int g    = blockIdx.y;
    const int d0   = blockIdx.x * 16;
    const int hi   = lane >> 4;
    const int mrow = lane & 15;
    const int dcol = d0 + mrow;
    const int pos  = posp[0];
    const int jbase = blockIdx.z * (CTX / JSPLIT);
    const _Float16* prow = probs + (size_t)(g * NREP + (mrow & (NREP - 1))) * CTX;
    const _Float16 rm = (mrow < NREP) ? (_Float16)1.0f : (_Float16)0.0f;
    const float vnewval = v_new[g * HD + dcol];      // hoisted: select, not load
    const float* Vcol = Vcache + (size_t)g * CTX * HD + dcol;
    v8f c = {};
    for (int j0 = jbase; j0 < jbase + CTX / JSPLIT; j0 += 32) {
        const int base = j0 + hi * 8;
        v8h alo = *(const v8h*)(prow + base) * rm;
        v8h ahi = *(const v8h*)(prow + base + 16) * rm;
        v16h a, b;
#pragma unroll
        for (int e = 0; e < 8; ++e) { a[e] = alo[e]; a[8 + e] = ahi[e]; }
#pragma unroll
        for (int e = 0; e < 16; ++e) {
            const int kk = ((e < 8) ? (base + e) : (base + 16 + (e - 8)));
            float vv = Vcol[(size_t)kk * HD];
            vv = (kk == pos) ? vnewval : vv;
            b[e] = (_Float16)vv;
        }
        c = __builtin_amdgcn_wmma_f32_16x16x32_f16(false, a, false, b,
                                                   (short)0, c, false, false);
    }
#pragma unroll
    for (int r = 0; r < 8; ++r) {
        const int M = r + 8 * hi;
        if (M < NREP)
            attn_part[(size_t)blockIdx.z * (NH * HD)
                      + (size_t)(g * NREP + M) * HD + dcol] = c[r];
    }
}

// ------------------------------------------------------------ elementwise ---
__global__ void gelu_mul_kernel(const float* __restrict__ g,
                                const float* __restrict__ u,
                                float* __restrict__ y, int n) {
    const int i = blockIdx.x * BLK + threadIdx.x;
    if (i < n) y[i] = gelu_tanh(g[i]) * u[i];
}

__global__ void copy_kernel(const float* __restrict__ src,
                            float* __restrict__ dst, int n) {
    const int i = blockIdx.x * BLK + threadIdx.x;
    if (i < n) dst[i] = src[i];
}

// ------------------------------------------------- argmax + tanh softcap ----
__global__ void argmax_softcap_kernel(const float* __restrict__ logits,
                                      float* __restrict__ out) {
    __shared__ float vmax[BLK];
    __shared__ int   vidx[BLK];
    float best = -1e30f; int bi = 0;
    for (int i = threadIdx.x; i < VOCAB; i += BLK) {
        float v = logits[i];
        if (v > best) { best = v; bi = i; }
    }
    vmax[threadIdx.x] = best; vidx[threadIdx.x] = bi; __syncthreads();
    for (int o = BLK / 2; o > 0; o >>= 1) {
        if (threadIdx.x < o) {
            const int t = threadIdx.x;
            if (vmax[t + o] > vmax[t] ||
                (vmax[t + o] == vmax[t] && vidx[t + o] < vidx[t])) {
                vmax[t] = vmax[t + o]; vidx[t] = vidx[t + o];
            }
        }
        __syncthreads();
    }
    if (threadIdx.x == 0) {
        out[0] = (float)vidx[0];                              // token_id
        out[1] = tanhf(vmax[0] / SOFTCAPF) * SOFTCAPF;        // token_logit
    }
}

// ---------------------------------------------------------------------------
extern "C" void kernel_launch(void* const* d_in, const int* in_sizes, int n_in,
                              void* d_out, int out_size, void* d_ws, size_t ws_size,
                              hipStream_t stream) {
    (void)in_sizes; (void)n_in; (void)out_size; (void)ws_size;

    // setup_inputs() dict order, params dict flattened in key insertion order.
    const float* inputs_embeds   = (const float*)d_in[0];
    const float* per_layer_input = (const float*)d_in[1];
    const float* kv_cache        = (const float*)d_in[2];  // [2L][NKV][CTX][HD]
    const float* w_in_ln         = (const float*)d_in[3];
    const float* w_post_attn_ln  = (const float*)d_in[4];
    const float* w_pre_ff_ln     = (const float*)d_in[5];
    const float* w_post_ff_ln    = (const float*)d_in[6];
    const float* w_post_pl_ln    = (const float*)d_in[7];
    const float* Wq              = (const float*)d_in[8];
    const float* Wk              = (const float*)d_in[9];
    const float* Wv              = (const float*)d_in[10];
    const float* Wo              = (const float*)d_in[11];
    const float* w_qnorm         = (const float*)d_in[12];
    const float* w_knorm         = (const float*)d_in[13];
    const float* Wg              = (const float*)d_in[14];
    const float* Wu              = (const float*)d_in[15];
    const float* Wd              = (const float*)d_in[16];
    const float* Wpl_gate        = (const float*)d_in[17];
    const float* Wpl_proj        = (const float*)d_in[18];
    const float* layer_scalar    = (const float*)d_in[19];
    const float* w_final         = (const float*)d_in[20];
    const float* Wlm             = (const float*)d_in[21];
    const float* cos_s           = (const float*)d_in[22];
    const float* sin_s           = (const float*)d_in[23];
    const float* cos_f           = (const float*)d_in[24];
    const float* sin_f           = (const float*)d_in[25];
    const int*   pos             = (const int*)d_in[26];
    const float* causal          = (const float*)d_in[27];
    // d_in[28] = update_mask (handled via j == pos substitution)

    // ---- workspace carve (all buffers fully written before read; ~1.5 MB)
    float* p = (float*)d_ws;
    auto alloc = [&](size_t n) { float* r = p; p += n; return r; };
    float*    h      = alloc(HDIM);
    float*    hn     = alloc(HDIM);
    float*    qbuf   = alloc(NH * HD);
    float*    kbuf   = alloc(NKV * HD);
    float*    vbuf   = alloc(NKV * HD);
    float*    k_new  = alloc(NKV * HD);
    float*    v_new  = alloc(NKV * HD);
    _Float16* qh     = (_Float16*)alloc(NH * HD / 2);
    float*    scores = alloc((size_t)NH * CTX);
    _Float16* probs  = (_Float16*)alloc((size_t)NH * CTX / 2);
    float*    attn   = alloc(NH * HD);
    float*    obuf   = alloc(HDIM);
    float*    gbuf   = alloc(FF);
    float*    ubuf   = alloc(FF);
    float*    mbuf   = alloc(FF);
    float*    plg    = alloc(PLD);
    float*    plo    = alloc(HDIM);
    float*    logits = alloc(VOCAB);
    float*    part   = alloc((size_t)SPLITK * VOCAB); // also AV partial tiles

    auto gemv = [&](const float* x, const float* W, float* y, int K, int N) {
        dim3 g((N + BLK - 1) / BLK, SPLITK);
        gemv_part_kernel<<<g, BLK, 0, stream>>>(x, W, part, K, N);
        gemv_reduce_kernel<<<(N + BLK - 1) / BLK, BLK, 0, stream>>>(part, y, N);
    };

    copy_kernel<<<HDIM / BLK, BLK, 0, stream>>>(inputs_embeds, h, HDIM);

    for (int l = 0; l < LAYERS; ++l) {
        const float* cosT = (l % 2 == 1) ? cos_f : cos_s;
        const float* sinT = (l % 2 == 1) ? sin_f : sin_s;
        const float* Kc = kv_cache + (size_t)l * NKV * CTX * HD;
        const float* Vc = kv_cache + (size_t)(LAYERS + l) * NKV * CTX * HD;

        // ---- attention block
        rmsnorm_kernel<<<1, BLK, 0, stream>>>(h, w_in_ln + (size_t)l * HDIM, hn, HDIM);
        gemv(hn, Wq + (size_t)l * HDIM * NH * HD, qbuf, HDIM, NH * HD);
        gemv(hn, Wk + (size_t)l * HDIM * NKV * HD, kbuf, HDIM, NKV * HD);
        gemv(hn, Wv + (size_t)l * HDIM * NKV * HD, vbuf, HDIM, NKV * HD);
        qkv_post_kernel<<<NH + 2 * NKV, HD, 0, stream>>>(
            qbuf, kbuf, vbuf, w_qnorm + (size_t)l * HD, w_knorm + (size_t)l * HD,
            cosT, sinT, pos, qh, k_new, v_new);
        attn_scores_kernel<<<dim3(CTX / 16, NKV), 32, 0, stream>>>(
            qh, Kc, k_new, causal, pos, scores);
        softmax_kernel<<<NH, BLK, 0, stream>>>(scores, probs);
        attn_av_kernel<<<dim3(HD / 16, NKV, JSPLIT), 32, 0, stream>>>(
            probs, Vc, v_new, pos, part);
        gemv_reduce_kernel<<<(NH * HD) / BLK, BLK, 0, stream>>>(part, attn, NH * HD);
        gemv(attn, Wo + (size_t)l * NH * HD * HDIM, obuf, NH * HD, HDIM);
        rms_add_kernel<<<1, BLK, 0, stream>>>(
            obuf, w_post_attn_ln + (size_t)l * HDIM, h, h, HDIM, nullptr, 0);

        // ---- feed-forward block
        rmsnorm_kernel<<<1, BLK, 0, stream>>>(h, w_pre_ff_ln + (size_t)l * HDIM, hn, HDIM);
        gemv(hn, Wg + (size_t)l * HDIM * FF, gbuf, HDIM, FF);
        gemv(hn, Wu + (size_t)l * HDIM * FF, ubuf, HDIM, FF);
        gelu_mul_kernel<<<FF / BLK, BLK, 0, stream>>>(gbuf, ubuf, mbuf, FF);
        gemv(mbuf, Wd + (size_t)l * FF * HDIM, obuf, FF, HDIM);
        rms_add_kernel<<<1, BLK, 0, stream>>>(
            obuf, w_post_ff_ln + (size_t)l * HDIM, h, h, HDIM, nullptr, 0);

        // ---- per-layer input block (+ layer scalar)
        gemv(h, Wpl_gate + (size_t)l * HDIM * PLD, plg, HDIM, PLD);
        gelu_mul_kernel<<<1, BLK, 0, stream>>>(
            plg, per_layer_input + (size_t)l * PLD, plg, PLD);
        gemv(plg, Wpl_proj + (size_t)l * PLD * HDIM, plo, PLD, HDIM);
        rms_add_kernel<<<1, BLK, 0, stream>>>(
            plo, w_post_pl_ln + (size_t)l * HDIM, h, h, HDIM, layer_scalar, l);
    }

    // ---- final norm + LM head + softcap + argmax
    rmsnorm_kernel<<<1, BLK, 0, stream>>>(h, w_final, hn, HDIM);
    gemv(hn, Wlm, logits, HDIM, VOCAB);
    argmax_softcap_kernel<<<1, BLK, 0, stream>>>(logits, (float*)d_out);
}
